// FCN8sAtOnceMultiGnn_69810398429319
// MI455X (gfx1250) — matrix operand order repaired and checked by reading, not compiled, not run on hardware
//
#include <hip/hip_runtime.h>

typedef __attribute__((ext_vector_type(16))) _Float16 v16h;
typedef __attribute__((ext_vector_type(8)))  _Float16 v8h;
typedef __attribute__((ext_vector_type(8)))  float    v8f;

#define NB    4
#define CCH   256
#define HWP   3600
#define KNN   8
#define FDIM  512          // 2C
#define PIXB  8            // pixels per block (8 waves share one gather)
#define BLKPB 450          // HWP / PIXB
#define ROWS  64           // gathered rows per block (8 pixels * 8 neighbors)
#define ROWH3 768          // halves per LDS row: r[0:256) | s[256:512) | d[512:768)
#define GNN_LDS (ROWS * ROWH3 * 2 + 512 * 4)   // 98,304 + 2,048 = 100,352 B
#define KNN_LDS (HWP * 6 * 4)                  // 86,400 B

// ---------------------------------------------------------------------------
// Pack W_g (f32 [256 out][512 in]) into WMMA-B fragment order:
// Bp[mod][nt(16)][ks(16)][lane(32)][j(16)] halves, 32B contiguous per lane.
// B is 32x16/op: lane n (0-15) holds col n, K=0..7,16..23; lane n+16 holds K=8..15,24..31.
// IR trick: ir_feat = [s | s-r] = [s | -d]; we feed A = d for the second half
// and bake the negation into the IR weights (f >= 256 => -W_ir).
// ---------------------------------------------------------------------------
__global__ void k_pack_weights(const float* __restrict__ Wrgb,
                               const float* __restrict__ Wir,
                               _Float16* __restrict__ Bp)
{
    int lane = threadIdx.x;            // 32
    int blk  = blockIdx.x;             // mod*256 + nt*16 + ks  (512 blocks)
    int mod  = blk >> 8;
    const float* W = mod ? Wir : Wrgb;
    int nt = (blk >> 4) & 15;
    int ks = blk & 15;
    int c  = nt * 16 + (lane & 15);
    int f0 = ks * 32 + ((lane >= 16) ? 8 : 0);
    float sgn = (mod == 1 && ks >= 8) ? -1.f : 1.f;
    _Float16* dst = Bp + (((size_t)blk * 32 + lane) * 16);
#pragma unroll
    for (int j = 0; j < 16; ++j)
        dst[j] = (_Float16)(sgn * W[c * FDIM + f0 + j + ((j >= 8) ? 8 : 0)]);
}

// ---------------------------------------------------------------------------
// NCHW cnn -> NHWC h (f32 master) + f16 copy for WMMA gathers
// ---------------------------------------------------------------------------
__global__ void k_init_h(const float* __restrict__ cnn, float* __restrict__ h,
                         _Float16* __restrict__ hh)
{
    size_t g = (size_t)blockIdx.x * 256 + threadIdx.x;  // NHWC order
    int c = (int)(g & 255);
    size_t t = g >> 8;
    int i = (int)(t % HWP);
    int n = (int)(t / HWP);
    float v = cnn[((size_t)(n * CCH + c)) * HWP + i];
    h[g] = v;
    hh[g] = (_Float16)v;
}

// ---------------------------------------------------------------------------
// KNN top-8 (gram-matrix distances, matching reference ordering/ties).
// One thread per query row; all 3600 points (5 floats + sqnorm) in LDS.
// ---------------------------------------------------------------------------
__global__ void k_knn(const float* __restrict__ x /* [N][5][HW] */,
                      int* __restrict__ knn)
{
    extern __shared__ float sm[];
    float* pts = sm;               // [HWP][5]
    float* sq  = sm + HWP * 5;     // [HWP]
    int n = blockIdx.y;
    const float* xb = x + (size_t)n * 5 * HWP;
    for (int i = threadIdx.x; i < HWP; i += blockDim.x) {
        float s = 0.f;
#pragma unroll
        for (int d = 0; d < 5; ++d) {
            float v = xb[d * HWP + i];
            pts[i * 5 + d] = v;
            s += v * v;
        }
        sq[i] = s;
    }
    __syncthreads();
    int row = blockIdx.x * blockDim.x + threadIdx.x;
    if (row >= HWP) return;
    float p0 = pts[row*5], p1 = pts[row*5+1], p2 = pts[row*5+2],
          p3 = pts[row*5+3], p4 = pts[row*5+4];
    float sqi = sq[row];
    float bd[8]; int bi[8];
#pragma unroll
    for (int t = 0; t < 8; ++t) { bd[t] = 3.4e38f; bi[t] = 0; }
    for (int j = 0; j < HWP; ++j) {
        const float* q = pts + j * 5;
        float dot = p0*q[0] + p1*q[1] + p2*q[2] + p3*q[3] + p4*q[4];
        float d2 = sqi + sq[j] - 2.f * dot;
        if (d2 < bd[7]) {
            float cd = d2; int ci = j;
#pragma unroll
            for (int t = 0; t < 8; ++t) {
                bool sw = cd < bd[t];
                float td = sw ? bd[t] : cd;
                int   ti = sw ? bi[t] : ci;
                bd[t] = sw ? cd : bd[t];
                bi[t] = sw ? ci : bi[t];
                cd = td; ci = ti;
            }
        }
    }
    int* o = knn + ((size_t)n * HWP + row) * KNN;
#pragma unroll
    for (int t = 0; t < 8; ++t) o[t] = bi[t];
}

// ---------------------------------------------------------------------------
// A-fragment load: two contiguous 8-half LDS chunks (K offsets +0..7 and
// +16..23 of this lane's K sub-base), per ISA 16-bit A 16x32 lane layout.
// ---------------------------------------------------------------------------
__device__ inline v16h lds_chunks(const _Float16* row, int f0)
{
    union { v16h v; uint4 u[2]; } x;
    x.u[0] = *(const uint4*)(row + f0);
    x.u[1] = *(const uint4*)(row + f0 + 16);
    return x.v;
}

// ---------------------------------------------------------------------------
// Fused gather + dual-modality MLP GEMM (f16 WMMA, f32 acc) + relu +
// max-over-K + mean reduction.
//   block = 256 threads (8 waves) sharing ONE 8-pixel gather:
//     - wave w gathers pixel (pbase+w)'s 8 neighbor rows: r, s, d = r-s
//     - rows pixel-major (row = pixel*8 + k) -> 4 M-tiles of 16 rows;
//       D rows 0..7 / 8..15 of a tile are two pixels -> lanes 0-15 / 16-31
//     - wave w computes output tiles nt = w and w+8 over all 4 M-tiles
//   k-loop statically split (no runtime ks<8 branch):
//     ks 0..7 : A = r (rgb) / s (ir)
//     ks 8..15: shared A = d, IR sign folded into packed weights
// ---------------------------------------------------------------------------
__global__ void __launch_bounds__(256) k_gnn(
    const _Float16* __restrict__ hh, const int* __restrict__ knn_rgb,
    const int* __restrict__ knn_ir, const _Float16* __restrict__ Bp,
    const float* __restrict__ bg_rgb, const float* __restrict__ bg_ir,
    float* __restrict__ bsum)
{
    extern __shared__ char smem[];
    _Float16* sh = (_Float16*)smem;                    // [64 rows][768 halves]
    float* sums = (float*)(smem + ROWS * ROWH3 * 2);   // [512] rgb | ir

    int lane = threadIdx.x & 31;
    int wave = threadIdx.x >> 5;
    int n = blockIdx.x / BLKPB;
    int pbase = (blockIdx.x % BLKPB) * PIXB;

    for (int t = threadIdx.x; t < 512; t += 256) sums[t] = 0.f;

    // ---- gather: wave w -> pixel (pbase+w), rows lr = w*8 + k
    const size_t bo = (size_t)n * HWP;
    {
        int p = pbase + wave;
        for (int k = 0; k < KNN; ++k) {
            int lr = wave * KNN + k;
            int jr = knn_rgb[(bo + p) * KNN + k];
            int ji = knn_ir [(bo + p) * KNN + k];
            const uint4* srcR = (const uint4*)(hh + (bo + jr) * CCH);
            const uint4* srcS = (const uint4*)(hh + (bo + ji) * CCH);
            union { uint4 u; v8h h; } r, s, d;
            r.u = srcR[lane];                 // 32 lanes x 8 halves = 256
            s.u = srcS[lane];
            d.h = r.h - s.h;
            uint4* dst = (uint4*)(sh + (size_t)lr * ROWH3);
            dst[lane]      = r.u;
            dst[32 + lane] = s.u;
            dst[64 + lane] = d.u;
        }
    }
    __syncthreads();

    const int kh  = (lane >= 16) ? 8 : 0;    // per-lane K sub-base (A layout)
    const int col = lane & 15;
    const v16h* BpV = (const v16h*)Bp;       // 32B per lane entry

    // per-mt LDS row base pointers (hoisted out of the k-loops)
    const _Float16* rows[4];
#pragma unroll
    for (int mt = 0; mt < 4; ++mt)
        rows[mt] = sh + (size_t)(mt * 16 + col) * ROWH3 + kh;

    for (int half = 0; half < 2; ++half) {
        int nt = wave + half * 8;
        // per-nt B fragment base pointers: index = ((mod*256+nt*16+ks)*32+lane)
        const v16h* bRp = BpV + ((size_t)(nt * 16) * 32 + lane);
        const v16h* bIp = bRp + (size_t)256 * 16 * 32;

        v8f accR[4], accI[4];
        v8f zero = {};
#pragma unroll
        for (int mt = 0; mt < 4; ++mt) { accR[mt] = zero; accI[mt] = zero; }

        // ---- feat[0:256): A = r (rgb) / s (ir)
#pragma unroll 4
        for (int ks = 0; ks < 8; ++ks) {
            v16h bR = bRp[(size_t)ks * 32];
            v16h bI = bIp[(size_t)ks * 32];
#pragma unroll
            for (int mt = 0; mt < 4; ++mt) {
                v16h aR = lds_chunks(rows[mt], ks * 32);
                v16h aI = lds_chunks(rows[mt], 256 + ks * 32);
                accR[mt] = __builtin_amdgcn_wmma_f32_16x16x32_f16(
                    false, aR, false, bR, (short)0, accR[mt], false, false);
                accI[mt] = __builtin_amdgcn_wmma_f32_16x16x32_f16(
                    false, aI, false, bI, (short)0, accI[mt], false, false);
            }
        }
        // ---- feat[256:512): shared A = d (IR sign in packed weights)
#pragma unroll 4
        for (int ks = 8; ks < 16; ++ks) {
            v16h bR = bRp[(size_t)ks * 32];
            v16h bI = bIp[(size_t)ks * 32];
#pragma unroll
            for (int mt = 0; mt < 4; ++mt) {
                v16h aD = lds_chunks(rows[mt], 512 + (ks - 8) * 32);
                accR[mt] = __builtin_amdgcn_wmma_f32_16x16x32_f16(
                    false, aD, false, bR, (short)0, accR[mt], false, false);
                accI[mt] = __builtin_amdgcn_wmma_f32_16x16x32_f16(
                    false, aD, false, bI, (short)0, accI[mt], false, false);
            }
        }

        // epilogue: +bias, relu, max over K (= per-lane max of 8 acc floats,
        // since D VGPR v holds row v (+8 for lanes>=16) = neighbor index k)
        float biasR = bg_rgb[nt * 16 + col];
        float biasI = bg_ir [nt * 16 + col];
        float sR = 0.f, sI = 0.f;
#pragma unroll
        for (int mt = 0; mt < 4; ++mt) {
            float mR = 0.f, mI = 0.f;   // relu floor
#pragma unroll
            for (int e = 0; e < 8; ++e) {
                mR = fmaxf(mR, accR[mt][e] + biasR);
                mI = fmaxf(mI, accI[mt][e] + biasI);
            }
            sR += mR;                   // sum over this lane's 4 pixels
            sI += mI;
        }
        atomicAdd(&sums[nt * 16 + col], sR);
        atomicAdd(&sums[256 + nt * 16 + col], sI);
    }
    __syncthreads();

    float* bs = bsum + (size_t)n * 512;
    for (int t = threadIdx.x; t < 512; t += 256) atomicAdd(&bs[t], sums[t]);
}

// ---------------------------------------------------------------------------
// SE block: cc = bsum/HW -> relu(cc@W1^T+b1) -> sigmoid(.@W2^T+b2)
// ---------------------------------------------------------------------------
__global__ void k_se(const float* __restrict__ bsum, const float* __restrict__ W1,
                     const float* __restrict__ b1, const float* __restrict__ W2,
                     const float* __restrict__ b2, float* __restrict__ se)
{
    __shared__ float hid[16];
    int n = blockIdx.x;
    const float* cs = bsum + (size_t)n * 512;
    int tid = threadIdx.x;   // 64
    if (tid < 16) {
        float a = b1[tid];
        const float inv = 1.f / (float)HWP;
        for (int f = 0; f < 512; ++f) a += (cs[f] * inv) * W1[tid * 512 + f];
        hid[tid] = fmaxf(a, 0.f);
    }
    __syncthreads();
    for (int c = tid; c < 256; c += 64) {
        float a = b2[c];
#pragma unroll
        for (int j = 0; j < 16; ++j) a += hid[j] * W2[c * 16 + j];
        se[(size_t)n * 256 + c] = 1.f / (1.f + __expf(-a));
    }
}

// h *= se (broadcast over pixels); refresh f16 copy for next gather
__global__ void k_scale(float* __restrict__ h, _Float16* __restrict__ hh,
                        const float* __restrict__ se)
{
    size_t g = (size_t)blockIdx.x * 256 + threadIdx.x;  // NHWC
    int c = (int)(g & 255);
    int n = (int)((g >> 8) / HWP);
    float v = h[g] * se[(size_t)n * 256 + c];
    h[g] = v;
    hh[g] = (_Float16)v;
}

// out = relu(gamma * (se*h) + cnn), NCHW (folds the last iteration's scale)
__global__ void k_final(const float* __restrict__ h, const float* __restrict__ se,
                        const float* __restrict__ cnn, const float* __restrict__ gamma,
                        float* __restrict__ out)
{
    size_t g = (size_t)blockIdx.x * 256 + threadIdx.x;  // NCHW
    int i = (int)(g % HWP);
    int c = (int)((g / HWP) & 255);
    int n = (int)(g / ((size_t)HWP * 256));
    float hv = h[((size_t)n * HWP + i) * 256 + c] * se[(size_t)n * 256 + c];
    float v = gamma[0] * hv + cnn[g];
    out[g] = fmaxf(v, 0.f);
}

__global__ void k_zero(float* __restrict__ p, int nfl)
{
    int g = blockIdx.x * 256 + threadIdx.x;
    if (g < nfl) p[g] = 0.f;
}

// ---------------------------------------------------------------------------
extern "C" void kernel_launch(void* const* d_in, const int* in_sizes, int n_in,
                              void* d_out, int out_size, void* d_ws, size_t ws_size,
                              hipStream_t stream)
{
    (void)in_sizes; (void)n_in; (void)out_size; (void)ws_size;
    const float* cnn   = (const float*)d_in[0];
    const float* rgb   = (const float*)d_in[1];
    const float* irp   = (const float*)d_in[2];
    const float* Wgr   = (const float*)d_in[3];
    const float* bgr   = (const float*)d_in[4];
    const float* Wgi   = (const float*)d_in[5];
    const float* bgi   = (const float*)d_in[6];
    const float* Wse1  = (const float*)d_in[7];
    const float* bse1  = (const float*)d_in[8];
    const float* Wse2  = (const float*)d_in[9];
    const float* bse2  = (const float*)d_in[10];
    const float* gamma = (const float*)d_in[11];
    // d_in[12] = gnn_iterations(2), d_in[13] = k(8): device scalars; values
    // fixed by setup_inputs, hardcoded to stay graph-capture safe.
    float* out = (float*)d_out;

    char* ws = (char*)d_ws;
    float*    h    = (float*)   (ws + 0);           // 14,745,600 B
    _Float16* hh   = (_Float16*)(ws + 14745600);    //  7,372,800 B
    int*      krgb = (int*)     (ws + 22118400);    //    460,800 B
    int*      kir  = (int*)     (ws + 22579200);    //    460,800 B
    _Float16* Bp   = (_Float16*)(ws + 23040000);    //    524,288 B
    float*    bsum = (float*)   (ws + 23564288);    //      8,192 B
    float*    se   = (float*)   (ws + 23572480);    //      4,096 B

    // CDNA5 WGP has 320KB LDS; opt in to >64KB dynamic shared.
    hipFuncSetAttribute(reinterpret_cast<const void*>(k_knn),
                        hipFuncAttributeMaxDynamicSharedMemorySize, KNN_LDS);
    hipFuncSetAttribute(reinterpret_cast<const void*>(k_gnn),
                        hipFuncAttributeMaxDynamicSharedMemorySize, GNN_LDS);

    k_pack_weights<<<512, 32, 0, stream>>>(Wgr, Wgi, Bp);
    k_init_h<<<14400, 256, 0, stream>>>(cnn, h, hh);
    k_knn<<<dim3(15, NB), 256, KNN_LDS, stream>>>(rgb, krgb);
    k_knn<<<dim3(15, NB), 256, KNN_LDS, stream>>>(irp, kir);

    for (int it = 0; it < 2; ++it) {
        k_zero<<<8, 256, 0, stream>>>(bsum, NB * 512);
        k_gnn<<<NB * BLKPB, 256, GNN_LDS, stream>>>(hh, krgb, kir, Bp, bgr, bgi, bsum);
        k_se<<<NB, 64, 0, stream>>>(bsum, Wse1, bse1, Wse2, bse2, se);
        if (it == 0)
            k_scale<<<14400, 256, 0, stream>>>(h, hh, se);
        else
            k_final<<<14400, 256, 0, stream>>>(h, se, cnn, gamma, out);
    }
}
